// MoELayer_24240795419274
// MI455X (gfx1250) — matrix operand order repaired and checked by reading, not compiled, not run on hardware
//
#include <hip/hip_runtime.h>
#include <hip/hip_bf16.h>
#include <stdint.h>

#define N_TOK 4096
#define DIM   1024
#define NEXP  8
#define HID   2048
#define TWOH  4096
#define LDT   40      // padded LDS stride in halves: 80B rows, 16B-aligned, 20-bank skew

typedef __attribute__((ext_vector_type(16))) __bf16 bf16x16;
typedef __attribute__((ext_vector_type(8)))  float  f32x8;

union FragAB {
    bf16x16 v;
    uint4   q[2];
};

__device__ __forceinline__ uint16_t f2bf(float f) {
    union { float f; uint32_t u; } c;
    c.f = f;
    uint32_t r = c.u + 0x7FFFu + ((c.u >> 16) & 1u);   // round-to-nearest-even
    return (uint16_t)(r >> 16);
}

// --- gfx1250 async global->LDS copy (ASYNCcnt-tracked), GVS addressing -----
__device__ __forceinline__ void async_b128(uint32_t lds_off, uint32_t voff, uint64_t sbase) {
    asm volatile("global_load_async_to_lds_b128 %0, %1, %2"
                 :: "v"(lds_off), "v"(voff), "s"(sbase) : "memory");
}
#define WAIT_ASYNC0() asm volatile("s_wait_asynccnt 0x0" ::: "memory")

// ---------------------------------------------------------------------------
// Elementwise f32 -> bf16 (x). 4 elements/thread, float4 in, uint2 out.
// ---------------------------------------------------------------------------
__global__ __launch_bounds__(256) void conv_ew(const float* __restrict__ in,
                                               uint16_t* __restrict__ out) {
    const int idx = blockIdx.x * 256 + threadIdx.x;
    const float4 v = ((const float4*)in)[idx];
    uint2 o;
    o.x = (uint32_t)f2bf(v.x) | ((uint32_t)f2bf(v.y) << 16);
    o.y = (uint32_t)f2bf(v.z) | ((uint32_t)f2bf(v.w) << 16);
    ((uint2*)out)[idx] = o;
}

// ---------------------------------------------------------------------------
// Per-expert transpose + convert: in [R][C] f32 -> out [C][R] bf16.
// blockIdx.z = expert, 32x32 tiles through LDS, coalesced both sides.
// ---------------------------------------------------------------------------
__global__ __launch_bounds__(256) void transpose_conv(const float* __restrict__ in,
                                                      uint16_t* __restrict__ out,
                                                      int R, int C) {
    __shared__ uint16_t tile[32][33];
    const size_t base = (size_t)blockIdx.z * R * C;
    const int c0 = blockIdx.x * 32;
    const int r0 = blockIdx.y * 32;
    const int tx = threadIdx.x, ty = threadIdx.y;
#pragma unroll
    for (int i = 0; i < 4; i++) {
        const int r = r0 + ty + i * 8;
        tile[ty + i * 8][tx] = f2bf(in[base + (size_t)r * C + c0 + tx]);
    }
    __syncthreads();
#pragma unroll
    for (int i = 0; i < 4; i++) {
        const int c = c0 + ty + i * 8;
        out[base + (size_t)c * R + r0 + tx] = tile[tx][ty + i * 8];
    }
}

// ---------------------------------------------------------------------------
// Router: one wave per token. logits = x @ gate_w^T, softmax, top-2, renorm.
// ---------------------------------------------------------------------------
__global__ __launch_bounds__(256) void moe_router(const float* __restrict__ X,
                                                  const float* __restrict__ GW,
                                                  float* __restrict__ LOGITS,
                                                  float* __restrict__ COMB) {
    const int lane = threadIdx.x & 31;
    const int wave = threadIdx.x >> 5;
    const int n = blockIdx.x * 8 + wave;
    const float* xr = X + (size_t)n * DIM;

    float acc[NEXP];
#pragma unroll
    for (int e = 0; e < NEXP; e++) acc[e] = 0.0f;

    for (int d = lane; d < DIM; d += 32) {
        const float xv = xr[d];
#pragma unroll
        for (int e = 0; e < NEXP; e++) acc[e] += xv * GW[e * DIM + d];
    }
#pragma unroll
    for (int e = 0; e < NEXP; e++) {
#pragma unroll
        for (int s = 16; s > 0; s >>= 1) acc[e] += __shfl_xor(acc[e], s, 32);
    }

    if (lane == 0) {
        float mx = acc[0];
#pragma unroll
        for (int e = 1; e < NEXP; e++) mx = fmaxf(mx, acc[e]);
        float p[NEXP];
        float sum = 0.0f;
#pragma unroll
        for (int e = 0; e < NEXP; e++) { p[e] = __expf(acc[e] - mx); sum += p[e]; }
        const float inv = 1.0f / sum;
#pragma unroll
        for (int e = 0; e < NEXP; e++) p[e] *= inv;

        int i0 = 0;
#pragma unroll
        for (int e = 1; e < NEXP; e++) if (p[e] > p[i0]) i0 = e;
        int i1 = (i0 == 0) ? 1 : 0;
#pragma unroll
        for (int e = 0; e < NEXP; e++) if (e != i0 && p[e] > p[i1]) i1 = e;

        const float wsum = p[i0] + p[i1];
        const float w0 = p[i0] / wsum;
        const float w1 = p[i1] / wsum;
#pragma unroll
        for (int e = 0; e < NEXP; e++) {
            COMB[(size_t)n * NEXP + e] = (e == i0) ? w0 : ((e == i1) ? w1 : 0.0f);
            LOGITS[(size_t)n * NEXP + e] = acc[e];
        }
    }
}

// ---------------------------------------------------------------------------
// GEMM1 + SwiGLU:  h = x @ w1[e];  act = silu(h[:,:H]) * h[:,H:]  (bf16 out)
// All operands pre-converted bf16; w1t is [2H][D] (transposed) so both A and
// B tiles stage as LINEAR async copies. Double-buffered LDS, async-to-LDS.
// WG: 128 rows x 128 cols of both halves; 8 waves (4x2); wave = 32x64/half.
// ---------------------------------------------------------------------------
__global__ __launch_bounds__(256) void moe_gemm1(const uint16_t* __restrict__ XB,
                                                 const uint16_t* __restrict__ W1T,
                                                 uint16_t* __restrict__ ACT,
                                                 int e) {
    __shared__ __align__(16) uint16_t As[2][128 * LDT];
    __shared__ __align__(16) uint16_t Bg[2][128 * LDT];
    __shared__ __align__(16) uint16_t Bl[2][128 * LDT];

    const uint16_t* W1e = W1T + (size_t)e * TWOH * DIM;   // [4096][1024] bf16
    const int tid  = threadIdx.x;
    const int lane = tid & 31;
    const int wave = tid >> 5;
    const int wm = wave & 3;
    const int wn = wave >> 2;
    const int rb = blockIdx.y * 128;
    const int cb = blockIdx.x * 128;       // gate-half column base, [0, HID)
    const int g  = lane >> 4;
    const int nl = lane & 15;

    f32x8 accg[2][4], accl[2][4];
#pragma unroll
    for (int mi = 0; mi < 2; mi++)
#pragma unroll
        for (int ni = 0; ni < 4; ni++)
#pragma unroll
            for (int r = 0; r < 8; r++) { accg[mi][ni][r] = 0.0f; accl[mi][ni][r] = 0.0f; }

    // --- async stage of one K-slab (32 halves deep): 6 b128 per thread ---
    auto stage = [&](int buf, int kb) {
        const uint64_t xbase = (uint64_t)(uintptr_t)(XB + kb);
        const uint64_t gbase = (uint64_t)(uintptr_t)(W1e + kb);
        const uint64_t lbase = (uint64_t)(uintptr_t)(W1e + (size_t)HID * DIM + kb);
#pragma unroll
        for (int j = 0; j < 2; j++) {
            const int c   = tid + j * 256;        // chunk id, 8 halves each
            const int row = c >> 2;
            const int kp  = (c & 3) * 8;
            const uint32_t lds_a = (uint32_t)(uintptr_t)&As[buf][row * LDT + kp];
            const uint32_t lds_g = (uint32_t)(uintptr_t)&Bg[buf][row * LDT + kp];
            const uint32_t lds_l = (uint32_t)(uintptr_t)&Bl[buf][row * LDT + kp];
            const uint32_t voff_a = (uint32_t)(((rb + row) * DIM + kp) * 2);
            const uint32_t voff_b = (uint32_t)(((cb + row) * DIM + kp) * 2);
            async_b128(lds_a, voff_a, xbase);
            async_b128(lds_g, voff_b, gbase);
            async_b128(lds_l, voff_b, lbase);
        }
    };

    stage(0, 0);
    int buf = 0;
    for (int kb = 0; kb < DIM; kb += 32) {
        WAIT_ASYNC0();          // this wave's chunks of current slab landed
        __syncthreads();        // all waves' chunks landed
        if (kb + 32 < DIM) stage(buf ^ 1, kb + 32);   // overlaps compute below

        FragAB a[2], bgf[4], blf[4];
#pragma unroll
        for (int mi = 0; mi < 2; mi++) {
            const uint16_t* p = &As[buf][(wm * 32 + mi * 16 + nl) * LDT];
            a[mi].q[0] = *(const uint4*)(p + g * 16);
            a[mi].q[1] = *(const uint4*)(p + g * 16 + 8);
        }
#pragma unroll
        for (int ni = 0; ni < 4; ni++) {
            const uint16_t* pg = &Bg[buf][(wn * 64 + ni * 16 + nl) * LDT + g * 16];
            bgf[ni].q[0] = *(const uint4*)(pg);
            bgf[ni].q[1] = *(const uint4*)(pg + 8);
            const uint16_t* pl = &Bl[buf][(wn * 64 + ni * 16 + nl) * LDT + g * 16];
            blf[ni].q[0] = *(const uint4*)(pl);
            blf[ni].q[1] = *(const uint4*)(pl + 8);
        }
#pragma unroll
        for (int mi = 0; mi < 2; mi++)
#pragma unroll
            for (int ni = 0; ni < 4; ni++) {
                accg[mi][ni] = __builtin_amdgcn_wmma_f32_16x16x32_bf16(
                    false, a[mi].v, false, bgf[ni].v, (short)0, accg[mi][ni], false, false);
                accl[mi][ni] = __builtin_amdgcn_wmma_f32_16x16x32_bf16(
                    false, a[mi].v, false, blf[ni].v, (short)0, accl[mi][ni], false, false);
            }
        __syncthreads();        // everyone done reading buf before it is restaged
        buf ^= 1;
    }

    // Epilogue: act = silu(gate) * lin, store bf16
#pragma unroll
    for (int mi = 0; mi < 2; mi++)
#pragma unroll
        for (int ni = 0; ni < 4; ni++)
#pragma unroll
            for (int r = 0; r < 8; r++) {
                const float gv = accg[mi][ni][r];
                const float lv = accl[mi][ni][r];
                const float s  = gv * __builtin_amdgcn_rcpf(1.0f + __expf(-gv));
                const int row = rb + wm * 32 + mi * 16 + g * 8 + r;
                const int col = cb + wn * 64 + ni * 16 + nl;
                ACT[(size_t)row * HID + col] = f2bf(s * lv);
            }
}

// ---------------------------------------------------------------------------
// GEMM2:  out += comb[:, e] * (act @ w2[e]);  w2t is [D][H] (transposed) bf16
// ---------------------------------------------------------------------------
__global__ __launch_bounds__(256) void moe_gemm2(const uint16_t* __restrict__ ACT,
                                                 const uint16_t* __restrict__ W2T,
                                                 const float* __restrict__ COMB,
                                                 float* __restrict__ OUT,
                                                 int e) {
    __shared__ __align__(16) uint16_t As[2][128 * LDT];
    __shared__ __align__(16) uint16_t Bt[2][128 * LDT];

    const uint16_t* W2e = W2T + (size_t)e * DIM * HID;    // [1024][2048] bf16
    const int tid  = threadIdx.x;
    const int lane = tid & 31;
    const int wave = tid >> 5;
    const int wm = wave & 3;
    const int wn = wave >> 2;
    const int rb = blockIdx.y * 128;
    const int cb = blockIdx.x * 128;
    const int g  = lane >> 4;
    const int nl = lane & 15;

    f32x8 acc[2][4];
#pragma unroll
    for (int mi = 0; mi < 2; mi++)
#pragma unroll
        for (int ni = 0; ni < 4; ni++)
#pragma unroll
            for (int r = 0; r < 8; r++) acc[mi][ni][r] = 0.0f;

    auto stage = [&](int buf, int kb) {
        const uint64_t abase = (uint64_t)(uintptr_t)(ACT + kb);
        const uint64_t bbase = (uint64_t)(uintptr_t)(W2e + kb);
#pragma unroll
        for (int j = 0; j < 2; j++) {
            const int c   = tid + j * 256;
            const int row = c >> 2;
            const int kp  = (c & 3) * 8;
            const uint32_t lds_a = (uint32_t)(uintptr_t)&As[buf][row * LDT + kp];
            const uint32_t lds_b = (uint32_t)(uintptr_t)&Bt[buf][row * LDT + kp];
            const uint32_t voff_a = (uint32_t)(((rb + row) * HID + kp) * 2);
            const uint32_t voff_b = (uint32_t)(((cb + row) * HID + kp) * 2);
            async_b128(lds_a, voff_a, abase);
            async_b128(lds_b, voff_b, bbase);
        }
    };

    stage(0, 0);
    int buf = 0;
    for (int kb = 0; kb < HID; kb += 32) {
        WAIT_ASYNC0();
        __syncthreads();
        if (kb + 32 < HID) stage(buf ^ 1, kb + 32);

        FragAB a[2], b[4];
#pragma unroll
        for (int mi = 0; mi < 2; mi++) {
            const uint16_t* p = &As[buf][(wm * 32 + mi * 16 + nl) * LDT];
            a[mi].q[0] = *(const uint4*)(p + g * 16);
            a[mi].q[1] = *(const uint4*)(p + g * 16 + 8);
        }
#pragma unroll
        for (int ni = 0; ni < 4; ni++) {
            const uint16_t* p = &Bt[buf][(wn * 64 + ni * 16 + nl) * LDT + g * 16];
            b[ni].q[0] = *(const uint4*)(p);
            b[ni].q[1] = *(const uint4*)(p + 8);
        }
#pragma unroll
        for (int mi = 0; mi < 2; mi++)
#pragma unroll
            for (int ni = 0; ni < 4; ni++)
                acc[mi][ni] = __builtin_amdgcn_wmma_f32_16x16x32_bf16(
                    false, a[mi].v, false, b[ni].v, (short)0, acc[mi][ni], false, false);
        __syncthreads();
        buf ^= 1;
    }

    // Epilogue: out[row, col] += comb[row, e] * acc   (experts run sequentially)
#pragma unroll
    for (int mi = 0; mi < 2; mi++)
#pragma unroll
        for (int r = 0; r < 8; r++) {
            const int row = rb + wm * 32 + mi * 16 + g * 8 + r;
            const float sc = COMB[(size_t)row * NEXP + e];
#pragma unroll
            for (int ni = 0; ni < 4; ni++) {
                const int col = cb + wn * 64 + ni * 16 + nl;
                OUT[(size_t)row * DIM + col] += sc * acc[mi][ni][r];
            }
        }
}

// ---------------------------------------------------------------------------
extern "C" void kernel_launch(void* const* d_in, const int* in_sizes, int n_in,
                              void* d_out, int out_size, void* d_ws, size_t ws_size,
                              hipStream_t stream) {
    const float* x  = (const float*)d_in[0];   // [4096, 1024]
    const float* gw = (const float*)d_in[1];   // [8, 1024]
    const float* w1 = (const float*)d_in[2];   // [8, 1024, 4096]
    const float* w2 = (const float*)d_in[3];   // [8, 2048, 1024]

    float* out    = (float*)d_out;                       // [4096*1024]
    float* logits = out + (size_t)N_TOK * DIM;           // [4096*8]

    uint16_t* xb  = (uint16_t*)d_ws;                     //  8 MB  [4096][1024]
    uint16_t* w1t = xb  + (size_t)N_TOK * DIM;           // 64 MB  [8][4096][1024]
    uint16_t* w2t = w1t + (size_t)NEXP * TWOH * DIM;     // 32 MB  [8][1024][2048]
    uint16_t* act = w2t + (size_t)NEXP * DIM * HID;      // 16 MB  [4096][2048]
    float* comb   = (float*)(act + (size_t)N_TOK * HID); // 128 KB [4096][8]

    hipMemsetAsync(out, 0, (size_t)N_TOK * DIM * sizeof(float), stream);

    // One-time bf16 conversion / transposition (graph-safe, ~20us of HBM time)
    conv_ew<<<(N_TOK * DIM) / (256 * 4), 256, 0, stream>>>(x, xb);
    transpose_conv<<<dim3(TWOH / 32, DIM / 32, NEXP), dim3(32, 8), 0, stream>>>(w1, w1t, DIM, TWOH);
    transpose_conv<<<dim3(DIM / 32, HID / 32, NEXP), dim3(32, 8), 0, stream>>>(w2, w2t, HID, DIM);

    moe_router<<<N_TOK / 8, 256, 0, stream>>>(x, gw, logits, comb);

    for (int e = 0; e < NEXP; e++) {
        moe_gemm1<<<dim3(HID / 128, N_TOK / 128), 256, 0, stream>>>(xb, w1t, act, e);
        moe_gemm2<<<dim3(DIM / 128, N_TOK / 128), 256, 0, stream>>>(act, w2t, comb, out, e);
    }
}
